// UnpadBmm1_3848290697289
// MI455X (gfx1250) — matrix operand order repaired and checked by reading, not compiled, not run on hardware
//
#include <hip/hip_runtime.h>

typedef __attribute__((ext_vector_type(16))) __bf16 v16bf;
typedef __attribute__((ext_vector_type(8)))  float  v8f;

#define HIDDEN 2048
#define HEADS  16
#define HSZ    128        // head_size = K depth
#define BT     64         // block tile edge (M and N)
#define LDS_K  136        // padded bf16 row stride: 272B -> bank-conflict-free
#define LDS_KU (LDS_K/2)  // row stride in dwords (68)

union FragCast { uint4 q[2]; v16bf v; };

__device__ __forceinline__ unsigned pack2_bf16(float x, float y) {
  // Native f32 -> bf16 converts (RNE); compiler emits hardware cvt on gfx1250
  // (bf16 is a first-class CDNA5 type) instead of ~6 integer VALU ops.
  union { __bf16 h[2]; unsigned u; } p;
  p.h[0] = (__bf16)x;
  p.h[1] = (__bf16)y;
  return p.u;
}

template <int S>
__global__ __launch_bounds__(128) void qkt_wmma_bf16(
    const float* __restrict__ A,   // seq base: [S, HEADS*HSZ]
    const float* __restrict__ B,   // seq base: [S, HEADS*HSZ]
    float* __restrict__ C)         // seq base: [HEADS, S, S]
{
  __shared__ unsigned int Abuf[BT * LDS_KU];
  __shared__ unsigned int Bbuf[BT * LDS_KU];

  const int h   = blockIdx.z;
  const int m0  = blockIdx.x * BT;
  const int n0  = blockIdx.y * BT;
  const int tid = threadIdx.x;

  // ---- stage A/B 64x128 tiles into LDS, converting fp32 -> bf16 ----
  // 128 threads: 32 threads per row (one float4 each), 4 rows per pass, 16 passes.
  {
    const int rb = tid >> 5;            // starting row 0..3
    const int k4 = (tid & 31) * 4;      // fixed element column per thread
    const float* ap = &A[(size_t)(m0 + rb) * HIDDEN + h * HSZ + k4];
    const float* bp = &B[(size_t)(n0 + rb) * HIDDEN + h * HSZ + k4];
    unsigned int* ao = &Abuf[rb * LDS_KU + (k4 >> 1)];
    unsigned int* bo = &Bbuf[rb * LDS_KU + (k4 >> 1)];
    #pragma unroll
    for (int it = 0; it < BT / 4; ++it) {
      const float4 av = *(const float4*)(ap + (size_t)it * 4 * HIDDEN);
      const float4 bv = *(const float4*)(bp + (size_t)it * 4 * HIDDEN);
      ao[it * 4 * LDS_KU]     = pack2_bf16(av.x, av.y);
      ao[it * 4 * LDS_KU + 1] = pack2_bf16(av.z, av.w);
      bo[it * 4 * LDS_KU]     = pack2_bf16(bv.x, bv.y);
      bo[it * 4 * LDS_KU + 1] = pack2_bf16(bv.z, bv.w);
    }
  }
  __syncthreads();

  const int wave   = tid >> 5;
  const int lane   = tid & 31;
  const int lanelo = lane & 15;
  const int hihalf = lane >> 4;        // 0 for lanes 0-15, 1 for 16-31
  const int wm = (wave >> 1) * 32;     // wave row base within block tile
  const int wn = (wave & 1)  * 32;     // wave col base within block tile

  v8f zero = {};
  v8f acc[2][2];
  acc[0][0] = zero; acc[0][1] = zero; acc[1][0] = zero; acc[1][1] = zero;

  #pragma unroll
  for (int kk = 0; kk < HSZ; kk += 32) {
    v16bf af[2], bfr[2];
    // A fragment (16x32 bf16): lane M = lanelo; K chunks kb..kb+7 & kb+16..+23, kb = hihalf*8
    #pragma unroll
    for (int t = 0; t < 2; ++t) {
      int m  = wm + t * 16 + lanelo;
      int kb = hihalf * 8;
      FragCast f;
      f.q[0] = *(const uint4*)&Abuf[m * LDS_KU + ((kk + kb) >> 1)];
      f.q[1] = *(const uint4*)&Abuf[m * LDS_KU + ((kk + 16 + kb) >> 1)];
      af[t] = f.v;
    }
    // B fragment (32x16 bf16): lane N = lanelo; K chunks kb..kb+7 & kb+16..+23, kb = hihalf*8
    #pragma unroll
    for (int t = 0; t < 2; ++t) {
      int n  = wn + t * 16 + lanelo;
      int kb = hihalf * 8;
      FragCast f;
      f.q[0] = *(const uint4*)&Bbuf[n * LDS_KU + ((kk + kb) >> 1)];
      f.q[1] = *(const uint4*)&Bbuf[n * LDS_KU + ((kk + 16 + kb) >> 1)];
      bfr[t] = f.v;
    }
    #pragma unroll
    for (int i = 0; i < 2; ++i)
      #pragma unroll
      for (int j = 0; j < 2; ++j)
        acc[i][j] = __builtin_amdgcn_wmma_f32_16x16x32_bf16(
            /*neg_a=*/false, af[i], /*neg_b=*/false, bfr[j],
            /*c_mod=*/(short)0, acc[i][j],
            /*reuse_a=*/false, /*reuse_b=*/false);
  }

  // ---- store C: f32 16x16 C/D layout: VGPR e -> M = e (+8 for hi half), N = lanelo ----
  // S is compile-time: row stride e*S*4 bytes folds into the store immediate offset.
  // Output is write-once, never re-read: non-temporal stores keep it out of L2.
  float* cbase = C + (size_t)h * (S * S);
  #pragma unroll
  for (int i = 0; i < 2; ++i) {
    #pragma unroll
    for (int j = 0; j < 2; ++j) {
      const unsigned gm = (unsigned)(m0 + wm + i * 16 + hihalf * 8);
      const unsigned gn = (unsigned)(n0 + wn + j * 16 + lanelo);
      float* p = cbase + (size_t)(gm * (unsigned)S + gn);
      #pragma unroll
      for (int e = 0; e < 8; ++e)
        __builtin_nontemporal_store(acc[i][j][e], p + e * S);
    }
  }
}

template <int S>
static inline void launch_seq(const float* a, const float* b, float* out,
                              hipStream_t stream) {
  dim3 grid(S / BT, S / BT, HEADS);
  qkt_wmma_bf16<S><<<grid, dim3(128), 0, stream>>>(a, b, out);
}

extern "C" void kernel_launch(void* const* d_in, const int* in_sizes, int n_in,
                              void* d_out, int out_size, void* d_ws, size_t ws_size,
                              hipStream_t stream) {
  (void)in_sizes; (void)n_in; (void)out_size; (void)d_ws; (void)ws_size;
  // Static seqlens from the reference (SEQLEN is a compile-time constant there).
  static const int SEQ[8] = {1024, 768, 512, 1536, 256, 1280, 512, 1024};

  const float* a = (const float*)d_in[0];
  const float* b = (const float*)d_in[1];
  float* out = (float*)d_out;

  size_t rowoff = 0, outoff = 0;
  for (int i = 0; i < 8; ++i) {
    const int s = SEQ[i];
    const float* ap = a + rowoff * HIDDEN;
    const float* bp = b + rowoff * HIDDEN;
    float* op = out + outoff;
    switch (s) {
      case 256:  launch_seq<256>(ap, bp, op, stream);  break;
      case 512:  launch_seq<512>(ap, bp, op, stream);  break;
      case 768:  launch_seq<768>(ap, bp, op, stream);  break;
      case 1024: launch_seq<1024>(ap, bp, op, stream); break;
      case 1280: launch_seq<1280>(ap, bp, op, stream); break;
      case 1536: launch_seq<1536>(ap, bp, op, stream); break;
      default: break;
    }
    rowoff += (size_t)s;
    outoff += (size_t)HEADS * (size_t)s * (size_t)s;
  }
}
